// FlashMHAttention_25701084299722
// MI455X (gfx1250) — matrix operand order repaired and checked by reading, not compiled, not run on hardware
//
#include <hip/hip_runtime.h>
#include <hip/hip_bf16.h>

// MI455X / gfx1250, wave32. Matmuls via v_wmma_f32_16x16x32_bf16.
// Tile staging global->LDS via the Tensor Data Mover (tensor_load_to_lds).
typedef __attribute__((ext_vector_type(16))) __bf16 v16bf;
typedef __attribute__((ext_vector_type(8)))  float  v8f;
typedef __attribute__((ext_vector_type(4)))  unsigned int u32x4;
typedef __attribute__((ext_vector_type(8)))  unsigned int u32x8;

#define BQ  2
#define LQ  2048
#define DQ  1024
#define HQ  16
#define DHQ 64

union AF { uint4 q[2]; v16bf v; };

__device__ __forceinline__ v8f zero8() {
  v8f z = {0.f,0.f,0.f,0.f,0.f,0.f,0.f,0.f};
  return z;
}

__device__ __forceinline__ v8f wmma_bf16(v16bf a, v16bf b, v8f c) {
  // (neg_a, A, neg_b, B, c_mod, C, reuse_a, reuse_b)
  return __builtin_amdgcn_wmma_f32_16x16x32_bf16(false, a, false, b, (short)0, c, false, false);
}

// low 32 bits of a generic LDS address = byte offset in LDS aperture
__device__ __forceinline__ unsigned lds_off(const void* p) {
  return (unsigned)(unsigned long long)(uintptr_t)p;
}

// ---------------------------------------------------------------------------
// TDM: DMA a 2D tile of bf16 (data_size=2B) from global memory into LDS.
// Descriptor per CDNA5 ISA section 8 (D# groups 0/1); 2D tensor -> two SGPR
// groups (VADDR2/VADDR3 = NULL). Issue once per wave; tracked by TENSORcnt.
//   tileD0  : tile width  (elements, fastest dim, contiguous in LDS)
//   tileD1  : tile height (rows)
//   strideD0: row stride in the global tensor (elements)
// ---------------------------------------------------------------------------
__device__ __forceinline__ void tdm_load_2d_bf16(unsigned ldsAddr, const void* gptr,
                                                 unsigned tileD0, unsigned tileD1,
                                                 unsigned long long strideD0,
                                                 unsigned tensorD0, unsigned tensorD1) {
  const unsigned long long ga = (unsigned long long)(uintptr_t)gptr;
  u32x4 g0;
  g0.x = 1u;                                                  // count=1, user D#
  g0.y = ldsAddr;                                             // lds_addr (bytes)
  g0.z = (unsigned)ga;                                        // global_addr[31:0]
  g0.w = (unsigned)((ga >> 32) & 0x01ffffffu) | (2u << 30);   // addr[56:32] | type=2
  u32x8 g1;
  g1[0] = 1u << 16;                                           // data_size=1 (2B), mask=0
  g1[1] = (tensorD0 & 0xffffu) << 16;                         // tensor_dim0[15:0]
  g1[2] = (tensorD0 >> 16) | ((tensorD1 & 0xffffu) << 16);    // dim0[31:16] | dim1[15:0]
  g1[3] = (tensorD1 >> 16) | (tileD0 << 16);                  // dim1[31:16] | tile_dim0
  g1[4] = tileD1 & 0xffffu;                                   // tile_dim1, tile_dim2=0
  g1[5] = (unsigned)strideD0;                                 // dim0_stride[31:0]
  g1[6] = (unsigned)((strideD0 >> 32) & 0xffffu);             // dim0_stride[47:32]
  g1[7] = 0u;
  asm volatile("tensor_load_to_lds %0, %1" :: "s"(g0), "s"(g1) : "memory");
}

// ---------------------------------------------------------------------------
// f32 -> bf16 streaming conversion
// ---------------------------------------------------------------------------
__global__ __launch_bounds__(256)
void cvt_f32_bf16(const float* __restrict__ src, __bf16* __restrict__ dst, int n) {
  int i = (blockIdx.x * blockDim.x + threadIdx.x) * 4;
  if (i < n) {
    float4 f = *(const float4*)(src + i);
    dst[i + 0] = (__bf16)f.x;
    dst[i + 1] = (__bf16)f.y;
    dst[i + 2] = (__bf16)f.z;
    dst[i + 3] = (__bf16)f.w;
  }
}

// ---------------------------------------------------------------------------
// C = A @ B^T   (A:[M][K] bf16 row-major, B:[N][K] bf16 row-major, K%32==0)
// block = 256 threads (8 waves), tile 128(M) x 64(N); wave tile 32x32.
// K-slices staged via TDM, double-buffered (DMA of slice k+1 overlaps the
// WMMAs of slice k).
// mode 0: store f32 to Cout[M][N]
// mode 1: QKV epilogue -> split-heads bf16 Q(scaled by 1/8)/K/V buffers
// ---------------------------------------------------------------------------
__global__ __launch_bounds__(256)
void gemm_bf16_nt(const __bf16* __restrict__ A, const __bf16* __restrict__ Bm,
                  int M, int N, int K,
                  float* __restrict__ Cout, int mode,
                  __bf16* __restrict__ qh, __bf16* __restrict__ kh,
                  __bf16* __restrict__ vh) {
  __shared__ alignas(16) __bf16 At[2][128 * 32];
  __shared__ alignas(16) __bf16 Bt[2][64 * 32];

  const int t    = threadIdx.x;
  const int lane = t & 31;
  const int wave = t >> 5;        // 0..7
  const int wm   = wave & 3;      // 0..3 : 32-row band
  const int wn   = wave >> 2;     // 0..1 : 32-col band
  const int r15  = lane & 15;
  const int hi   = lane >> 4;     // lane-half
  const int m0   = blockIdx.x * 128;
  const int n0   = blockIdx.y * 64;

  v8f acc[2][2] = {{zero8(), zero8()}, {zero8(), zero8()}};

  const __bf16* Abase = A  + (size_t)m0 * K;   // tile (m0, 0)
  const __bf16* Bbase = Bm + (size_t)n0 * K;   // tile (n0, 0)
  const int nk = K / 32;

  // prologue: DMA slice 0 into buffer 0
  if (wave == 0) {
    tdm_load_2d_bf16(lds_off(&At[0][0]), Abase, 32, 128,
                     (unsigned long long)K, (unsigned)K, (unsigned)M);
    tdm_load_2d_bf16(lds_off(&Bt[0][0]), Bbase, 32, 64,
                     (unsigned long long)K, (unsigned)K, (unsigned)N);
    __builtin_amdgcn_s_wait_tensorcnt(0);
  }
  __syncthreads();

  for (int kk = 0; kk < nk; ++kk) {
    const int buf = kk & 1;
    // kick off the next slice's DMA while computing this one
    if (wave == 0 && kk + 1 < nk) {
      tdm_load_2d_bf16(lds_off(&At[buf ^ 1][0]), Abase + (kk + 1) * 32, 32, 128,
                       (unsigned long long)K, (unsigned)K, (unsigned)M);
      tdm_load_2d_bf16(lds_off(&Bt[buf ^ 1][0]), Bbase + (kk + 1) * 32, 32, 64,
                       (unsigned long long)K, (unsigned)K, (unsigned)N);
    }

    // A fragments: 16x32, ISA layout (row=lane&15, K half-split by lane-half)
    AF a[2], b[2];
#pragma unroll
    for (int i = 0; i < 2; ++i) {
      const __bf16* p = &At[buf][(wm * 32 + i * 16 + r15) * 32 + hi * 8];
      a[i].q[0] = *(const uint4*)p;
      a[i].q[1] = *(const uint4*)(p + 16);
    }
    // B fragments: 32x16 (col=lane&15, K half-split by lane-half, same as A)
#pragma unroll
    for (int j = 0; j < 2; ++j) {
      const __bf16* p = &Bt[buf][(wn * 32 + j * 16 + r15) * 32 + hi * 8];
      b[j].q[0] = *(const uint4*)p;
      b[j].q[1] = *(const uint4*)(p + 16);
    }
#pragma unroll
    for (int i = 0; i < 2; ++i)
#pragma unroll
      for (int j = 0; j < 2; ++j)
        acc[i][j] = wmma_bf16(a[i].v, b[j].v, acc[i][j]);

    if (wave == 0 && kk + 1 < nk) __builtin_amdgcn_s_wait_tensorcnt(0);
    __syncthreads();
  }

  // epilogue: C layout = row (VGPR r + 8*lanehalf), col (lane&15)
#pragma unroll
  for (int i = 0; i < 2; ++i)
#pragma unroll
    for (int j = 0; j < 2; ++j) {
      const int ng = n0 + wn * 32 + j * 16 + r15;
#pragma unroll
      for (int r = 0; r < 8; ++r) {
        const int mg = m0 + wm * 32 + i * 16 + r + hi * 8;
        const float val = acc[i][j][r];
        if (mode == 0) {
          Cout[(size_t)mg * N + ng] = val;
        } else {
          // qkv split-heads: ng in [0,3072): which = q/k/v, head, dh
          const int which = ng >> 10;
          const int d     = ng & 1023;
          const int h     = d >> 6;
          const int dh    = d & 63;
          const int bb    = mg >> 11;      // / L
          const int l     = mg & 2047;     // % L
          const size_t dst = (((size_t)(bb * HQ + h)) * LQ + l) * DHQ + dh;
          // fold softmax scale 1/sqrt(64) = 0.125 into Q
          const __bf16 bv = (__bf16)(which == 0 ? val * 0.125f : val);
          __bf16* base = (which == 0) ? qh : (which == 1) ? kh : vh;
          base[dst] = bv;
        }
      }
    }
}

// ---------------------------------------------------------------------------
// Causal flash attention. grid = (L/64 q-tiles, B*H). block = 128 (4 waves).
// Each wave owns 16 q-rows; Br=64, Bc=64. K tile DMA'd via TDM ([lk][dh]);
// V staged transposed ([dh][lk]) by the VALU path concurrently with the DMA.
// ---------------------------------------------------------------------------
__global__ __launch_bounds__(128)
void flash_attn(const __bf16* __restrict__ Qh, const __bf16* __restrict__ Kh,
                const __bf16* __restrict__ Vh, __bf16* __restrict__ Ctx) {
  __shared__ alignas(16) __bf16 Kt[64 * 64];       // [lk][dh]  (TDM target)
  __shared__ alignas(16) __bf16 Vt[64 * 64];       // [dh][lk] (transposed)
  __shared__ alignas(16) __bf16 Pt[4 * 16 * 64];   // per-wave P, [row][lk]

  const int t    = threadIdx.x;
  const int lane = t & 31;
  const int wave = t >> 5;     // 0..3
  const int r15  = lane & 15;
  const int hi   = lane >> 4;
  const int qt   = blockIdx.x; // q tile (64 rows)
  const int bh   = blockIdx.y; // b*16 + h
  const size_t hbase = (size_t)bh * LQ * DHQ;
  const __bf16* Qb = Qh + hbase;
  const __bf16* Kb = Kh + hbase;
  const __bf16* Vb = Vh + hbase;

  const int q0 = qt * 64 + wave * 16;

  // Q fragments (scale already folded in) held in registers all kernel long.
  AF qf[2];
  {
    const __bf16* p = Qb + (size_t)(q0 + r15) * DHQ;
#pragma unroll
    for (int ks = 0; ks < 2; ++ks) {
      qf[ks].q[0] = *(const uint4*)(p + ks * 32 + hi * 8);
      qf[ks].q[1] = *(const uint4*)(p + ks * 32 + hi * 8 + 16);
    }
  }

  v8f O[4] = {zero8(), zero8(), zero8(), zero8()};
  float mI[8], lI[8];
#pragma unroll
  for (int r = 0; r < 8; ++r) { mI[r] = -INFINITY; lI[r] = 0.f; }

  const int srow = t >> 1;          // 0..63
  const int soff = (t & 1) * 32;    // 0 / 32

  for (int j = 0; j <= qt; ++j) {   // causal: skip upper-triangle tiles
    __syncthreads();                // previous iteration done with Kt/Vt
    // K tile: one TDM DMA (contiguous [64][64] bf16 rows)
    if (wave == 0) {
      tdm_load_2d_bf16(lds_off(&Kt[0]), Kb + (size_t)(j * 64) * DHQ,
                       64, 64, 64ull, 64u, (unsigned)LQ);
    }
    {   // stage V transposed (overlaps the K DMA)
      union { uint4 q[4]; __bf16 h[32]; } vv;
      const __bf16* src = Vb + (size_t)(j * 64 + srow) * DHQ + soff;
#pragma unroll
      for (int i = 0; i < 4; ++i) vv.q[i] = ((const uint4*)src)[i];
#pragma unroll
      for (int i = 0; i < 32; ++i) Vt[(soff + i) * 64 + srow] = vv.h[i];
    }
    if (wave == 0) __builtin_amdgcn_s_wait_tensorcnt(0);
    __syncthreads();

    // S = Q K^T : 4 column subtiles of 16, K-dim = DH = 64 (2 WMMAs each)
    v8f s[4];
#pragma unroll
    for (int sub = 0; sub < 4; ++sub) {
      AF bk0, bk1;
      const __bf16* p = &Kt[(sub * 16 + r15) * 64 + hi * 8];
      bk0.q[0] = *(const uint4*)p;        bk0.q[1] = *(const uint4*)(p + 16);
      bk1.q[0] = *(const uint4*)(p + 32); bk1.q[1] = *(const uint4*)(p + 48);
      v8f a = zero8();
      a = wmma_bf16(qf[0].v, bk0.v, a);
      a = wmma_bf16(qf[1].v, bk1.v, a);
      s[sub] = a;
    }

    if (j == qt) {  // diagonal tile: causal mask
#pragma unroll
      for (int sub = 0; sub < 4; ++sub) {
        const int kg = j * 64 + sub * 16 + r15;
#pragma unroll
        for (int r = 0; r < 8; ++r) {
          const int qg = q0 + r + hi * 8;
          if (kg > qg) s[sub][r] = -INFINITY;
        }
      }
    }

    // online softmax: each row spans 16 lanes of one lane-half
#pragma unroll
    for (int r = 0; r < 8; ++r) {
      float mx = fmaxf(fmaxf(s[0][r], s[1][r]), fmaxf(s[2][r], s[3][r]));
#pragma unroll
      for (int off = 8; off > 0; off >>= 1) mx = fmaxf(mx, __shfl_xor(mx, off, 16));
      const float mNew  = fmaxf(mI[r], mx);
      const float alpha = __expf(mI[r] - mNew);
      mI[r] = mNew;
      float rs = 0.f;
#pragma unroll
      for (int sub = 0; sub < 4; ++sub) {
        const float pv = __expf(s[sub][r] - mNew);
        s[sub][r] = pv;
        rs += pv;
      }
#pragma unroll
      for (int off = 8; off > 0; off >>= 1) rs += __shfl_xor(rs, off, 16);
      lI[r] = lI[r] * alpha + rs;
#pragma unroll
      for (int sub = 0; sub < 4; ++sub) O[sub][r] *= alpha;
    }

    // P -> LDS (C-layout -> A-layout transpose through LDS)
#pragma unroll
    for (int sub = 0; sub < 4; ++sub)
#pragma unroll
      for (int r = 0; r < 8; ++r)
        Pt[(wave * 16 + r + hi * 8) * 64 + sub * 16 + r15] = (__bf16)s[sub][r];
    __syncthreads();

    // O += P @ V
    AF pf[2];
    {
      const __bf16* p = &Pt[(wave * 16 + r15) * 64];
#pragma unroll
      for (int ks = 0; ks < 2; ++ks) {
        pf[ks].q[0] = *(const uint4*)(p + ks * 32 + hi * 8);
        pf[ks].q[1] = *(const uint4*)(p + ks * 32 + hi * 8 + 16);
      }
    }
#pragma unroll
    for (int sub = 0; sub < 4; ++sub) {
      AF bv0, bv1;
      const __bf16* p = &Vt[(sub * 16 + r15) * 64 + hi * 8];
      bv0.q[0] = *(const uint4*)p;        bv0.q[1] = *(const uint4*)(p + 16);
      bv1.q[0] = *(const uint4*)(p + 32); bv1.q[1] = *(const uint4*)(p + 48);
      O[sub] = wmma_bf16(pf[0].v, bv0.v, O[sub]);
      O[sub] = wmma_bf16(pf[1].v, bv1.v, O[sub]);
    }
  }

  // normalize, write ctx bf16 in [B*L][D] (col = h*64+dh) for the out-proj GEMM
  const int h  = bh & 15;
  const int bb = bh >> 4;
#pragma unroll
  for (int sub = 0; sub < 4; ++sub) {
    const int dh = sub * 16 + r15;
#pragma unroll
    for (int r = 0; r < 8; ++r) {
      const int l = q0 + r + hi * 8;
      const float v = O[sub][r] / lI[r];
      Ctx[((size_t)(bb * LQ + l)) * DQ + h * DHQ + dh] = (__bf16)v;
    }
  }
}

// ---------------------------------------------------------------------------
extern "C" void kernel_launch(void* const* d_in, const int* in_sizes, int n_in,
                              void* d_out, int out_size, void* d_ws, size_t ws_size,
                              hipStream_t stream) {
  const float* x    = (const float*)d_in[0];
  const float* Wqkv = (const float*)d_in[1];
  const float* Wout = (const float*)d_in[2];
  // d_in[3] = causal flag (config is always causal=1; kernels assume causal)

  const size_t nX    = (size_t)BQ * LQ * DQ;        // 4 Mi elems
  const size_t nWqkv = (size_t)3 * DQ * DQ;         // 3 Mi
  const size_t nWout = (size_t)DQ * DQ;             // 1 Mi
  const size_t nHead = (size_t)BQ * HQ * LQ * DHQ;  // 4 Mi

  char* ws = (char*)d_ws;                 // total 48 MiB of bf16 scratch
  __bf16* xb  = (__bf16*)ws; ws += nX * 2;
  __bf16* wqb = (__bf16*)ws; ws += nWqkv * 2;
  __bf16* wob = (__bf16*)ws; ws += nWout * 2;
  __bf16* qh  = (__bf16*)ws; ws += nHead * 2;
  __bf16* kh  = (__bf16*)ws; ws += nHead * 2;
  __bf16* vh  = (__bf16*)ws; ws += nHead * 2;
  __bf16* ctx = (__bf16*)ws; ws += nX * 2;

  cvt_f32_bf16<<<(int)(nX / 1024),    256, 0, stream>>>(x,    xb,  (int)nX);
  cvt_f32_bf16<<<(int)(nWqkv / 1024), 256, 0, stream>>>(Wqkv, wqb, (int)nWqkv);
  cvt_f32_bf16<<<(int)(nWout / 1024), 256, 0, stream>>>(Wout, wob, (int)nWout);

  // qkv = x @ Wqkv^T  -> split heads, scale Q by 0.125
  dim3 g1(4096 / 128, 3072 / 64);
  gemm_bf16_nt<<<g1, 256, 0, stream>>>(xb, wqb, 4096, 3072, 1024,
                                       nullptr, 1, qh, kh, vh);

  // causal flash attention
  dim3 g2(LQ / 64, BQ * HQ);
  flash_attn<<<g2, 128, 0, stream>>>(qh, kh, vh, ctx);

  // out = ctx @ Wout^T  (f32 output)
  dim3 g3(4096 / 128, 1024 / 64);
  gemm_bf16_nt<<<g3, 256, 0, stream>>>(ctx, wob, 4096, 1024, 1024,
                                       (float*)d_out, 0, nullptr, nullptr, nullptr);
}
